// DualStateLinearAttention_61984968016009
// MI455X (gfx1250) — matrix operand order repaired and checked by reading, not compiled, not run on hardware
//
#include <hip/hip_runtime.h>
#include <hip/hip_bf16.h>
#include <cstdint>
#include <cstddef>

// ---------------------------------------------------------------------------
// DualStateLinearAttention for MI455X (gfx1250, wave32, WMMA + TDM)
//   B=2, S=2048, HID=2048, H=16, D=128
//   1) convert hidden + weights to bf16
//   2) q,k,v (bf16 out) and g1,g2 (f32 out) = hidden @ W^T  [WMMA GEMM,
//      double-buffered LDS]
//   3) gate transform: clamp(logsigmoid(x+b)/16, -50)
//   4) chunked GLA scan (chunk=32): TDM double-buffers q/k/v/g chunk tiles in
//      LDS; all WMMA operands are stored so fragment loads are lane-contiguous
//      (ds_load_b128), incl. transposed state SbT, kT, vT
//   5) combine with softmax(alpha), final WMMA GEMM @ Wo^T -> d_out (fp32)
// ---------------------------------------------------------------------------

typedef __attribute__((ext_vector_type(16))) __bf16 v16bf;
typedef __attribute__((ext_vector_type(8)))  float  v8f;
typedef __attribute__((ext_vector_type(4)))  unsigned int v4u;
typedef __attribute__((ext_vector_type(8)))  int    v8i;
typedef __attribute__((ext_vector_type(4)))  int    v4i;

#define BATCH 2
#define SEQ   2048
#define HID   2048
#define NH    16
#define HD    128
#define CHUNK 32
#define NCHUNK (SEQ / CHUNK)

#define GEMM_TM 128
#define GEMM_TN 128
#define GEMM_TK 32

// ---- CDNA5 WMMA fragment loaders (wave32 layouts per ISA 7.12.2) ----------
// Both loaders read two contiguous 16-byte runs per lane -> 2x ds_load_b128.

// A 16x32 bf16 tile from row-major [m][k]; lane L: m=L&15, half=L>>4;
// VGPR j holds K = 16*(j/4) + 8*half + 2*(j%4).
__device__ __forceinline__ v16bf load_a_16x32(const __bf16* base, int ld) {
  const int lane = threadIdx.x & 31;
  const int m  = lane & 15;
  const int hh = (lane >> 4) & 1;
  v16bf a;
#pragma unroll
  for (int j = 0; j < 8; ++j) {
    const int kk = ((j >> 2) << 4) + (hh << 3) + ((j & 3) << 1);
    a[2 * j]     = base[m * ld + kk];
    a[2 * j + 1] = base[m * ld + kk + 1];
  }
  return a;
}

// B 32x16 bf16 tile, source stored [n][k] (B^T row-major); K = 16*(j/8) + 8*half + (j%8).
__device__ __forceinline__ v16bf load_b_nk(const __bf16* base, int ld) {
  const int lane = threadIdx.x & 31;
  const int n  = lane & 15;
  const int hh = (lane >> 4) & 1;
  v16bf b;
#pragma unroll
  for (int j = 0; j < 16; ++j) b[j] = base[n * ld + ((j >> 3) << 4) + (hh << 3) + (j & 7)];
  return b;
}

// ---- Tensor Data Mover: 2D tile load into LDS (D# per ISA 8.3/8.4) --------
// dsz_code: 1 = 2-byte elems (bf16), 2 = 4-byte elems (f32).
__device__ __forceinline__ void tdm_load_tile(unsigned lds_off, const void* gtile,
                                              unsigned dsz_code,
                                              unsigned tile_cols, unsigned tile_rows,
                                              unsigned row_stride,
                                              unsigned tensor_cols, unsigned tensor_rows) {
  const unsigned long long ga = (unsigned long long)(uintptr_t)gtile;
  v4u g0;
  g0[0] = 1u;                                   // count=1, user descriptor
  g0[1] = lds_off;                              // LDS byte address
  g0[2] = (unsigned)ga;                         // global_addr[31:0]
  g0[3] = ((unsigned)(ga >> 32) & 0x01ffffffu)  // global_addr[56:32]
        | (2u << 30);                           // type = 2 ("image")
  v8i g1;
  g1[0] = (int)(dsz_code << 16);                // wg_mask=0, data_size
  g1[1] = (int)((tensor_cols & 0xffffu) << 16);               // tensor_dim0 lo
  g1[2] = (int)((tensor_cols >> 16) | ((tensor_rows & 0xffffu) << 16));
  g1[3] = (int)((tensor_rows >> 16) | ((tile_cols & 0xffffu) << 16));  // tile_dim0
  g1[4] = (int)(tile_rows & 0xffffu);           // tile_dim1 (tile_dim2=0)
  g1[5] = (int)row_stride;                      // tensor_dim0_stride[31:0]
  g1[6] = 0;                                    // stride hi, dim1_stride lo
  g1[7] = 0;
  const v4i gz4 = {0, 0, 0, 0};                 // groups 2/3 unused (2D tile)
  const v8i gz8 = {0, 0, 0, 0, 0, 0, 0, 0};
  __builtin_amdgcn_tensor_load_to_lds(g0, g1, gz4, gz4, gz8, 0);
}

// ---- fp32 -> bf16 convert --------------------------------------------------
__global__ __launch_bounds__(256) void cvt_bf16(const float* __restrict__ in,
                                                __bf16* __restrict__ out, int n) {
  int i = blockIdx.x * blockDim.x + threadIdx.x;
  const int stride = gridDim.x * blockDim.x;
  for (; i < n; i += stride) out[i] = (__bf16)in[i];
}

// ---- C[M,N] = A[M,K] @ W[N,K]^T, bf16 in, f32 accumulate, OutT out ---------
template <typename OutT>
__global__ __launch_bounds__(256) void wmma_gemm_nt(const __bf16* __restrict__ A,
                                                    const __bf16* __restrict__ W,
                                                    OutT* __restrict__ C,
                                                    int M, int N, int K) {
  __shared__ __bf16 As[2][GEMM_TM][GEMM_TK];   // 2 x 8 KB
  __shared__ __bf16 Ws[2][GEMM_TN][GEMM_TK];   // 2 x 8 KB
  const int bm = blockIdx.y * GEMM_TM;
  const int bn = blockIdx.x * GEMM_TN;
  const int tid  = threadIdx.x;
  const int wave = tid >> 5;            // 8 waves: wave = m-tile row, 8 n-tiles
  const int lrow = tid >> 1;            // 0..127
  const int lcb  = (tid & 1) << 4;      // 0 or 16

  const __bf16* apBase = A + (size_t)(bm + lrow) * K + lcb;
  const __bf16* wpBase = W + (size_t)(bn + lrow) * K + lcb;

  v8f acc[8] = {};

  // preload tile 0
#pragma unroll
  for (int j = 0; j < 16; ++j) {
    As[0][lrow][lcb + j] = apBase[j];
    Ws[0][lrow][lcb + j] = wpBase[j];
  }
  __syncthreads();

  int buf = 0;
  for (int k0 = 0; k0 < K; k0 += GEMM_TK) {
    const bool more = (k0 + GEMM_TK) < K;
    __bf16 ra[16], rw[16];
    if (more) {
      const __bf16* ap = apBase + k0 + GEMM_TK;
      const __bf16* wp = wpBase + k0 + GEMM_TK;
      __builtin_prefetch(ap + GEMM_TK, 0, 1);
#pragma unroll
      for (int j = 0; j < 16; ++j) { ra[j] = ap[j]; rw[j] = wp[j]; }
    }

    v16bf af = load_a_16x32(&As[buf][wave * 16][0], GEMM_TK);
#pragma unroll
    for (int nt = 0; nt < 8; ++nt) {
      v16bf bv = load_b_nk(&Ws[buf][nt * 16][0], GEMM_TK);
      acc[nt] = __builtin_amdgcn_wmma_f32_16x16x32_bf16(
          false, af, false, bv, (short)0, acc[nt], false, false);
    }

    if (more) {
#pragma unroll
      for (int j = 0; j < 16; ++j) {
        As[buf ^ 1][lrow][lcb + j] = ra[j];
        Ws[buf ^ 1][lrow][lcb + j] = rw[j];
      }
    }
    __syncthreads();
    buf ^= 1;
  }

  const int lane = tid & 31;
  const int n  = lane & 15;
  const int hl = (lane >> 4) & 1;
#pragma unroll
  for (int nt = 0; nt < 8; ++nt) {
#pragma unroll
    for (int r = 0; r < 8; ++r) {
      C[(size_t)(bm + wave * 16 + r + 8 * hl) * N + bn + nt * 16 + n] =
          (OutT)acc[nt][r];
    }
  }
}

// ---- gate transform: g = max(logsigmoid(x + bias)/16, -50), in place -------
__global__ __launch_bounds__(256) void gate_transform(float* __restrict__ gbuf,
                                                      const float* __restrict__ bias,
                                                      int n) {
  const int i = blockIdx.x * blockDim.x + threadIdx.x;
  if (i >= n) return;
  const float x  = gbuf[i] + bias[i & (HID - 1)];
  const float ls = fminf(x, 0.0f) - log1pf(__expf(-fabsf(x)));
  gbuf[i] = fmaxf(ls * (1.0f / 16.0f), -50.0f);
}

// ---- chunked GLA scan: one block per (batch, head); TDM double buffering ---
__global__ __launch_bounds__(256) void gla_scan(const __bf16* __restrict__ qb,
                                                const __bf16* __restrict__ kb,
                                                const __bf16* __restrict__ vb,
                                                const float*  __restrict__ g,
                                                float* __restrict__ o) {
  __shared__ __bf16 stq[2][CHUNK][HD];   // TDM-staged q tiles   (2 x 8 KB)
  __shared__ __bf16 stk[2][CHUNK][HD];   // TDM-staged k tiles   (2 x 8 KB)
  __shared__ __bf16 stv[2][CHUNK][HD];   // TDM-staged v tiles   (2 x 8 KB)
  __shared__ float  stg[2][CHUNK][HD];   // TDM-staged gate tiles(2 x 16 KB)
  __shared__ __bf16 qs[CHUNK][HD];       // q~ = q * e^{G} * scale       [t][d]
  __shared__ __bf16 ks[CHUNK][HD];       // k~ = k * e^{-G}              [t][d]
  __shared__ __bf16 kT[HD][CHUNK];       // k~ transposed                [d][t]
  __shared__ __bf16 vT[HD][CHUNK];       // v transposed                 [d][t]
  __shared__ __bf16 As[CHUNK][CHUNK];    // tril(q~ k~^T)                [t][t']
  __shared__ __bf16 SbT[HD][HD];         // state transposed: SbT[n][k]=S[k][n]
  __shared__ float  Glast[HD];           // chunk-final gate cumsum
  __shared__ float  eGlast[HD];          // exp(Glast)

  const int bb   = blockIdx.x / NH;
  const int head = blockIdx.x % NH;
  const int tid  = threadIdx.x;
  const int wave = tid >> 5;
  const int lane = tid & 31;
  const int ln   = lane & 15;
  const int lh   = (lane >> 4) & 1;
  const float scale = 0.088388347648318447f;   // 128^-0.5

  for (int i = tid; i < HD * HD; i += 256) (&SbT[0][0])[i] = (__bf16)0.0f;

  const size_t rowbase = (size_t)bb * SEQ;
  const int col0 = head * HD;

  // waves 0..3 each own one tensor stream (q,k,v,g)
  auto stage_chunk = [&](int buf, int chunk) {
    const size_t roff = (rowbase + (size_t)chunk * CHUNK) * HID + col0;
    if (wave == 0)
      tdm_load_tile((unsigned)(uintptr_t)&stq[buf][0][0], qb + roff, 1u,
                    HD, CHUNK, HID, HID, BATCH * SEQ);
    else if (wave == 1)
      tdm_load_tile((unsigned)(uintptr_t)&stk[buf][0][0], kb + roff, 1u,
                    HD, CHUNK, HID, HID, BATCH * SEQ);
    else if (wave == 2)
      tdm_load_tile((unsigned)(uintptr_t)&stv[buf][0][0], vb + roff, 1u,
                    HD, CHUNK, HID, HID, BATCH * SEQ);
    else if (wave == 3)
      tdm_load_tile((unsigned)(uintptr_t)&stg[buf][0][0], g + roff, 2u,
                    HD, CHUNK, HID, HID, BATCH * SEQ);
  };

  if (wave < 4) stage_chunk(0, 0);   // prologue: stage chunk 0

  int cur = 0;
  for (int c = 0; c < NCHUNK; ++c, cur ^= 1) {
    // kick off chunk c+1, then wait until chunk c's tiles have landed
    if (wave < 4) {
      if (c + 1 < NCHUNK) {
        stage_chunk(cur ^ 1, c + 1);
        __builtin_amdgcn_s_wait_tensorcnt(1);
      } else {
        __builtin_amdgcn_s_wait_tensorcnt(0);
      }
    }
    __syncthreads();

    // Phase 1: per-column gate cumsum; gated bf16 tiles + transposed copies.
    // Thread d writes kT[d][*]/vT[d][*] rows contiguously (merges to b64/b128).
    if (tid < HD) {
      const int d = tid;
      float gacc = 0.0f;
#pragma unroll 4
      for (int t = 0; t < CHUNK; ++t) {
        gacc += stg[cur][t][d];
        const float eg = __expf(gacc);
        const __bf16 qv = (__bf16)((float)stq[cur][t][d] * eg * scale);
        const __bf16 kv = (__bf16)((float)stk[cur][t][d] * __expf(-gacc));
        const __bf16 vv = stv[cur][t][d];
        qs[t][d] = qv;
        ks[t][d] = kv;
        kT[d][t] = kv;
        vT[d][t] = vv;
      }
      Glast[d]  = gacc;
      eGlast[d] = __expf(gacc);
    }
    __syncthreads();

    // Phase 2: A = tril(q~ @ k~^T), 32x32, K=128 (waves 0..3, one tile each)
    if (wave < 4) {
      const int ti = wave >> 1, tj = wave & 1;
      v8f acc = {};
#pragma unroll
      for (int k0 = 0; k0 < HD; k0 += 32) {
        v16bf af = load_a_16x32(&qs[ti * 16][k0], HD);
        v16bf bv = load_b_nk(&ks[tj * 16][k0], HD);   // B[k=d][n=i] = ks[i][d]
        acc = __builtin_amdgcn_wmma_f32_16x16x32_bf16(
            false, af, false, bv, (short)0, acc, false, false);
      }
#pragma unroll
      for (int r = 0; r < 8; ++r) {
        const int mrow = ti * 16 + r + 8 * lh;
        const int ncol = tj * 16 + ln;
        As[mrow][ncol] = (__bf16)((ncol <= mrow) ? acc[r] : 0.0f);
      }
    }
    __syncthreads();

    // Phase 3: o = A @ V + q~ @ S_prev (32x128, 16 tiles, 2 per wave)
#pragma unroll
    for (int u = 0; u < 2; ++u) {
      const int idx = wave * 2 + u;
      const int ti = idx >> 3, tj = idx & 7;
      v8f acc = {};
      {
        v16bf af = load_a_16x32(&As[ti * 16][0], CHUNK);     // K = 32
        v16bf bv = load_b_nk(&vT[tj * 16][0], CHUNK);        // B[k=t][n=d]
        acc = __builtin_amdgcn_wmma_f32_16x16x32_bf16(
            false, af, false, bv, (short)0, acc, false, false);
      }
#pragma unroll
      for (int k0 = 0; k0 < HD; k0 += 32) {                  // K = 128
        v16bf af = load_a_16x32(&qs[ti * 16][k0], HD);
        v16bf bv = load_b_nk(&SbT[tj * 16][k0], HD);         // B[k][n]=SbT[n][k]
        acc = __builtin_amdgcn_wmma_f32_16x16x32_bf16(
            false, af, false, bv, (short)0, acc, false, false);
      }
#pragma unroll
      for (int r = 0; r < 8; ++r) {
        const int trow = c * CHUNK + ti * 16 + r + 8 * lh;
        o[(rowbase + trow) * HID + col0 + tj * 16 + ln] = acc[r];
      }
    }
    __syncthreads();   // all SbT reads for this chunk complete

    // Phase 4: S <- diag(e^{Glast}) (S + K~^T V); wave w owns d-rows w*16..+15
    // Per lane, the 8 state elements are consecutive bf16 in SbT[nc][*].
    {
      const int d0 = wave * 16;
      v16bf af = load_a_16x32(&kT[d0][0], CHUNK);   // A[m=d][k=t] = kT[d][t]
#pragma unroll
      for (int tj = 0; tj < 8; ++tj) {
        v8f acc = {};
        v16bf bv = load_b_nk(&vT[tj * 16][0], CHUNK);
        acc = __builtin_amdgcn_wmma_f32_16x16x32_bf16(
            false, af, false, bv, (short)0, acc, false, false);
        const int nc = tj * 16 + ln;
        const int db = d0 + 8 * lh;     // 8 consecutive d per lane
#pragma unroll
        for (int r = 0; r < 8; ++r) {
          const float prev = (float)SbT[nc][db + r];
          SbT[nc][db + r] = (__bf16)(eGlast[db + r] * (prev + acc[r]));
        }
      }
    }
    __syncthreads();
  }
}

// ---- combine: attn_bf16 = softmax(alpha)[0]*o1 + [1]*o2 --------------------
__global__ __launch_bounds__(256) void combine(const float* __restrict__ o1,
                                               const float* __restrict__ o2,
                                               const float* __restrict__ alpha,
                                               __bf16* __restrict__ attnb, int n) {
  const int i = blockIdx.x * blockDim.x + threadIdx.x;
  if (i >= n) return;
  const float a0 = alpha[0], a1 = alpha[1];
  const float m  = fmaxf(a0, a1);
  const float e0 = __expf(a0 - m), e1 = __expf(a1 - m);
  const float w0 = e0 / (e0 + e1);
  attnb[i] = (__bf16)(w0 * o1[i] + (1.0f - w0) * o2[i]);
}

// ---------------------------------------------------------------------------
extern "C" void kernel_launch(void* const* d_in, const int* in_sizes, int n_in,
                              void* d_out, int out_size, void* d_ws, size_t ws_size,
                              hipStream_t stream) {
  (void)in_sizes; (void)n_in; (void)out_size; (void)ws_size;

  const float* hidden = (const float*)d_in[0];
  const float* Wq  = (const float*)d_in[1];
  const float* Wk  = (const float*)d_in[2];
  const float* Wv  = (const float*)d_in[3];
  const float* Wo  = (const float*)d_in[4];
  const float* Wg1 = (const float*)d_in[5];
  const float* bg1 = (const float*)d_in[6];
  const float* Wg2 = (const float*)d_in[7];
  const float* bg2 = (const float*)d_in[8];
  const float* alpha = (const float*)d_in[9];

  const int M  = BATCH * SEQ;            // 4096
  const int MH = M * HID;                // 8,388,608
  const int WW = HID * HID;              // 4,194,304

  char* ws = (char*)d_ws;
  __bf16* hb    = (__bf16*)ws; ws += (size_t)MH * 2;
  __bf16* wqb   = (__bf16*)ws; ws += (size_t)WW * 2;
  __bf16* wkb   = (__bf16*)ws; ws += (size_t)WW * 2;
  __bf16* wvb   = (__bf16*)ws; ws += (size_t)WW * 2;
  __bf16* wob   = (__bf16*)ws; ws += (size_t)WW * 2;
  __bf16* wg1b  = (__bf16*)ws; ws += (size_t)WW * 2;
  __bf16* wg2b  = (__bf16*)ws; ws += (size_t)WW * 2;
  __bf16* qbuf  = (__bf16*)ws; ws += (size_t)MH * 2;
  __bf16* kbuf  = (__bf16*)ws; ws += (size_t)MH * 2;
  __bf16* vbuf  = (__bf16*)ws; ws += (size_t)MH * 2;
  float*  g1f   = (float*)ws;  ws += (size_t)MH * 4;
  float*  g2f   = (float*)ws;  ws += (size_t)MH * 4;
  float*  o1f   = (float*)ws;  ws += (size_t)MH * 4;
  float*  o2f   = (float*)ws;  ws += (size_t)MH * 4;
  __bf16* attnb = (__bf16*)ws; ws += (size_t)MH * 2;

  // 1) fp32 -> bf16
  cvt_bf16<<<2048, 256, 0, stream>>>(hidden, hb, MH);
  cvt_bf16<<<2048, 256, 0, stream>>>(Wq,  wqb,  WW);
  cvt_bf16<<<2048, 256, 0, stream>>>(Wk,  wkb,  WW);
  cvt_bf16<<<2048, 256, 0, stream>>>(Wv,  wvb,  WW);
  cvt_bf16<<<2048, 256, 0, stream>>>(Wo,  wob,  WW);
  cvt_bf16<<<2048, 256, 0, stream>>>(Wg1, wg1b, WW);
  cvt_bf16<<<2048, 256, 0, stream>>>(Wg2, wg2b, WW);

  // 2) projections (WMMA GEMMs): q,k,v in bf16; gates in f32
  dim3 ggrid(HID / GEMM_TN, M / GEMM_TM);   // (16, 32)
  wmma_gemm_nt<__bf16><<<ggrid, 256, 0, stream>>>(hb, wqb,  qbuf, M, HID, HID);
  wmma_gemm_nt<__bf16><<<ggrid, 256, 0, stream>>>(hb, wkb,  kbuf, M, HID, HID);
  wmma_gemm_nt<__bf16><<<ggrid, 256, 0, stream>>>(hb, wvb,  vbuf, M, HID, HID);
  wmma_gemm_nt<float ><<<ggrid, 256, 0, stream>>>(hb, wg1b, g1f,  M, HID, HID);
  wmma_gemm_nt<float ><<<ggrid, 256, 0, stream>>>(hb, wg2b, g2f,  M, HID, HID);

  // 3) gate transforms
  gate_transform<<<MH / 256, 256, 0, stream>>>(g1f, bg1, MH);
  gate_transform<<<MH / 256, 256, 0, stream>>>(g2f, bg2, MH);

  // 4) chunked GLA scans (one block per (b,h)), TDM-pipelined
  gla_scan<<<BATCH * NH, 256, 0, stream>>>(qbuf, kbuf, vbuf, g1f, o1f);
  gla_scan<<<BATCH * NH, 256, 0, stream>>>(qbuf, kbuf, vbuf, g2f, o2f);

  // 5) combine + output projection into d_out (fp32)
  combine<<<MH / 256, 256, 0, stream>>>(o1f, o2f, alpha, attnb, MH);
  wmma_gemm_nt<float><<<ggrid, 256, 0, stream>>>(attnb, wob, (float*)d_out,
                                                 M, HID, HID);
}
